// ReRankTransformer_71253507441103
// MI455X (gfx1250) — matrix operand order repaired and checked by reading, not compiled, not run on hardware
//
#include <hip/hip_runtime.h>
#include <hip/hip_bf16.h>

// Problem constants (from reference)
#define BB 32
#define MM 20000
#define NN_ 500000
#define CC 128
#define DD 128
#define HH 4
#define KK 100

typedef __attribute__((ext_vector_type(16))) _Float16 v16h;
typedef __attribute__((ext_vector_type(8)))  _Float16 v8h;
typedef __attribute__((ext_vector_type(8)))  float    v8f;

// ---------------------------------------------------------------------------
// WMMA fragment loaders (wave32, v_wmma_f32_16x16x32_f16)
// A (16x32, MxK), 16-bit: lanes 0-15 hold M=0-15 with K=0..15,
// lanes 16-31 hold the same rows with K=16..31 (matches B fragment).
// ---------------------------------------------------------------------------
__device__ __forceinline__ v16h load_a_frag(const _Float16* __restrict__ X,
                                            int m0, int k0, int lane) {
  const _Float16* row = X + (size_t)(m0 + (lane & 15)) * DD;
  int kA = k0 + ((lane < 16) ? 0 : 16);
  int kB = k0 + ((lane < 16) ? 8 : 24);
  v8h lo = *(const v8h*)(row + kA);
  v8h hi = *(const v8h*)(row + kB);
  v16h r;
#pragma unroll
  for (int e = 0; e < 8; ++e) { r[e] = lo[e]; r[e + 8] = hi[e]; }
  return r;
}

// B (32x16, KxN) built from an f32 row-major matrix holding B^T (i.e. W[n][k]):
// lane n=lane&15 is the column, lanes 0-15 carry K=0..15, lanes 16-31 K=16..31.
// f32 -> f16 conversion happens here (memory-bound kernels: free).
__device__ __forceinline__ v16h load_b_frag_f32(const float* __restrict__ Wm,
                                                int n0, int k0, int lane) {
  const float* p = Wm + (size_t)(n0 + (lane & 15)) * DD + k0 + ((lane < 16) ? 0 : 16);
  v16h r;
#pragma unroll
  for (int e = 0; e < 16; ++e) r[e] = (_Float16)p[e];
  return r;
}

// ---------------------------------------------------------------------------
// Generic Y[i,n] = sum_k Xf16[i,k] * W[n,k] (+bias[n]) (+rowOff[i]); K = 128.
// One wave computes a 32x16 output tile: two 16x16 WMMA accumulators sharing
// each B fragment (halves the f32->f16 conversion work on streamed W).
// Used for: [32x500000] logits GEMM and all five 128x128 projections.
// ---------------------------------------------------------------------------
__global__ void k_gemm_wmma(const _Float16* __restrict__ X, const float* __restrict__ Wm,
                            const float* __restrict__ bias, float* __restrict__ Y,
                            int I, int Nn, const float* __restrict__ rowOff) {
  int wave = blockIdx.x * (blockDim.x >> 5) + (threadIdx.x >> 5);
  int lane = threadIdx.x & 31;
  int Mg = I >> 5;                 // groups of 32 rows
  int Nt = (Nn + 15) >> 4;
  if (wave >= Mg * Nt) return;     // wave-uniform: EXEC stays all-ones for WMMA
  int mg = wave % Mg, nt = wave / Mg;
  int m0 = mg * 32, n0 = nt * 16;

  // Speculative prefetch of the next n-tile's weight rows (global_prefetch_b8)
  __builtin_prefetch(Wm + (size_t)(n0 + 16 + (lane & 15)) * DD, 0, 1);

  v8f acc0 = {}; v8f acc1 = {};
#pragma unroll
  for (int c = 0; c < 4; ++c) {
    int k0 = c * 32;
    v16h a0 = load_a_frag(X, m0, k0, lane);
    v16h a1 = load_a_frag(X, m0 + 16, k0, lane);
    v16h bf = load_b_frag_f32(Wm, n0, k0, lane);
    acc0 = __builtin_amdgcn_wmma_f32_16x16x32_f16(false, a0, false, bf, (short)0,
                                                  acc0, false, false);
    acc1 = __builtin_amdgcn_wmma_f32_16x16x32_f16(false, a1, false, bf, (short)0,
                                                  acc1, false, false);
  }
  int n = n0 + (lane & 15);
  if (n < Nn) {
    float bn = bias ? bias[n] : 0.0f;
    int ib = m0 + ((lane < 16) ? 0 : 8);
#pragma unroll
    for (int r = 0; r < 8; ++r) {
      float o0 = rowOff ? rowOff[ib + r] : 0.0f;
      float o1 = rowOff ? rowOff[ib + 16 + r] : 0.0f;
      Y[(size_t)(ib + r) * Nn + n]      = acc0[r] + bn + o0;
      Y[(size_t)(ib + 16 + r) * Nn + n] = acc1[r] + bn + o1;
    }
  }
}

// ---------------------------------------------------------------------------
// lhs_proj = lhs @ proj_W^T + proj_b  (32x128), plus per-row offsets
// off_emb_row[b] = lhs_proj[b] . off_emb_W + b0 ; off_id_row likewise.
// ---------------------------------------------------------------------------
__global__ void k_lhs_proj(const float* __restrict__ lhs, const float* __restrict__ projW,
                           const float* __restrict__ projb,
                           const float* __restrict__ offEW, const float* __restrict__ offEb,
                           const float* __restrict__ offIW, const float* __restrict__ offIb,
                           float* __restrict__ lhsProjF, _Float16* __restrict__ lhsProjH,
                           float* __restrict__ offEmbRow, float* __restrict__ offIdRow) {
  __shared__ float sp[BB][DD];
  int tid = threadIdx.x;
  for (int o = tid; o < BB * DD; o += blockDim.x) {
    int b = o / DD, d = o % DD;
    const float* a = lhs + (size_t)b * CC;
    const float* w = projW + (size_t)d * CC;
    float s = projb[d];
#pragma unroll 8
    for (int c = 0; c < CC; ++c) s += a[c] * w[c];
    sp[b][d] = s;
  }
  __syncthreads();
  for (int o = tid; o < BB * DD; o += blockDim.x) {
    int b = o / DD, d = o % DD;
    lhsProjF[o] = sp[b][d];
    lhsProjH[o] = (_Float16)sp[b][d];
  }
  if (tid < BB) {
    float s1 = offEb[0], s2 = offIb[0];
#pragma unroll 8
    for (int d = 0; d < DD; ++d) { s1 += sp[tid][d] * offEW[d]; s2 += sp[tid][d] * offIW[d]; }
    offEmbRow[tid] = s1;
    offIdRow[tid]  = s2;
  }
}

// ---------------------------------------------------------------------------
// idgnn logits on the M sampled nodes, scattered into the dense logits.
// ---------------------------------------------------------------------------
__global__ void k_idgnn(const float* __restrict__ lhs, const float* __restrict__ gnn,
                        const float* __restrict__ headW, const float* __restrict__ headb,
                        const float* __restrict__ offIdRow, const int* __restrict__ idx,
                        const int* __restrict__ batch, float* __restrict__ embgnn) {
  int m = blockIdx.x * blockDim.x + threadIdx.x;
  if (m >= MM) return;
  int b = batch[m];
  const float* gp = gnn + (size_t)m * CC;
  const float* lp = lhs + (size_t)b * CC;
  float s1 = 0.f, s2 = 0.f;
#pragma unroll 8
  for (int c = 0; c < CC; ++c) { float g = gp[c]; s1 += g * headW[c]; s2 += g * lp[c]; }
  embgnn[(size_t)b * NN_ + idx[m]] = s1 + headb[0] + s2 + offIdRow[b];
}

__global__ void k_fill_i32(int* __restrict__ p, int v, int n) {
  int i = blockIdx.x * blockDim.x + threadIdx.x;
  if (i < n) p[i] = v;
}
__global__ void k_fill_f32(float* __restrict__ p, float v, long long n) {
  long long i = (long long)blockIdx.x * blockDim.x + threadIdx.x;
  long long st = (long long)gridDim.x * blockDim.x;
  for (; i < n; i += st) p[i] = v;
}
__global__ void k_node2m_scatter(const int* __restrict__ idx, int* __restrict__ node2m) {
  int m = blockIdx.x * blockDim.x + threadIdx.x;
  if (m < MM) node2m[idx[m]] = m;
}

// ---------------------------------------------------------------------------
// Exact top-100 per row via 4-pass radix select over monotonic float-key bits.
// One block (1024 threads) per row; rows are L2-resident (2 MB each, 64 MB total
// vs 192 MB L2), so the 5 row sweeps are cheap. Tie-break matches jax.lax.top_k:
// value descending, index ascending.
// ---------------------------------------------------------------------------
__device__ __forceinline__ unsigned fkey(float f) {
  unsigned u = __float_as_uint(f);
  return u ^ ((u >> 31) ? 0xFFFFFFFFu : 0x80000000u);
}

#define TIE_CAP 1024
__global__ void k_topk(const float* __restrict__ logits, int* __restrict__ topk_out) {
  int b = blockIdx.x;
  const float* row = logits + (size_t)b * NN_;
  __shared__ unsigned hist[256];
  __shared__ unsigned s_prefix, s_need;
  __shared__ int s_nG, s_nE;
  __shared__ unsigned gkey[128];
  __shared__ int gidx[128];
  __shared__ int eidx[TIE_CAP];
  __shared__ unsigned skey[KK];
  __shared__ int sidx[KK];
  int tid = threadIdx.x, nthr = blockDim.x;

  if (tid == 0) { s_prefix = 0u; s_need = KK; }
  for (int p = 3; p >= 0; --p) {
    for (int i = tid; i < 256; i += nthr) hist[i] = 0u;
    __syncthreads();
    unsigned hm = (p == 3) ? 0u : (0xFFFFFFFFu << ((p + 1) * 8));
    unsigned pref = s_prefix;
    for (int i = tid; i < NN_; i += nthr) {
      unsigned u = fkey(row[i]);
      if ((u & hm) == pref) atomicAdd(&hist[(u >> (p * 8)) & 255], 1u);
    }
    __syncthreads();
    if (tid == 0) {
      unsigned need = s_need, cum = 0; int sel = 0;
      for (int bin = 255; bin >= 0; --bin) {
        if (cum + hist[bin] >= need) { sel = bin; break; }
        cum += hist[bin];
      }
      s_prefix = pref | ((unsigned)sel << (p * 8));
      s_need = need - cum;
    }
    __syncthreads();
  }
  unsigned T = s_prefix;
  int needEq = (int)s_need;                 // # of ==T elements to take
  if (tid == 0) { s_nG = 0; s_nE = 0; }
  __syncthreads();
  for (int i = tid; i < NN_; i += nthr) {
    unsigned u = fkey(row[i]);
    if (u > T) {
      int pos = atomicAdd(&s_nG, 1);
      if (pos < 128) { gkey[pos] = u; gidx[pos] = i; }
    } else if (u == T) {
      int pos = atomicAdd(&s_nE, 1);
      if (pos < TIE_CAP) eidx[pos] = i;
    }
  }
  __syncthreads();
  int nG = s_nG < KK ? s_nG : KK;           // strictly-greater count (<= 99)
  int nE = s_nE < TIE_CAP ? s_nE : TIE_CAP;
  for (int i = tid; i < nG; i += nthr) { skey[i] = gkey[i]; sidx[i] = gidx[i]; }
  // needEq smallest indices among the ==T set
  for (int t = tid; t < nE; t += nthr) {
    int v = eidx[t], r = 0;
    for (int j = 0; j < nE; ++j) r += (eidx[j] < v);
    if (r < needEq && nG + r < KK) { skey[nG + r] = T; sidx[nG + r] = v; }
  }
  __syncthreads();
  // final rank: value desc, index asc (unique ranks)
  for (int t = tid; t < KK; t += nthr) {
    unsigned kk = skey[t]; int ii = sidx[t]; int r = 0;
    for (int j = 0; j < KK; ++j) {
      unsigned kj = skey[j]; int ij = sidx[j];
      r += (kj > kk) || (kj == kk && ij < ii);
    }
    topk_out[b * KK + r] = sidx[t];
  }
}

// ---------------------------------------------------------------------------
// top_embed[b,t,:] = rhs_emb_weight[idx] + (node2m[idx]>=0 ? gnn[node2m[idx]] : 0)
// (final_rhs_embed never materialized). Writes f32 + f16 copies.
// ---------------------------------------------------------------------------
__global__ void k_gather(const float* __restrict__ W, const float* __restrict__ gnn,
                         const int* __restrict__ node2m, const int* __restrict__ topk,
                         float* __restrict__ teF, _Float16* __restrict__ teH) {
  int g = blockIdx.x * blockDim.x + threadIdx.x;
  if (g >= BB * KK * (DD / 4)) return;
  int rowi = g / (DD / 4);
  int e = (g % (DD / 4)) * 4;
  int idx = topk[rowi];
  int m = node2m[idx];
  float4 v = *(const float4*)(W + (size_t)idx * DD + e);
  if (m >= 0) {
    float4 gv = *(const float4*)(gnn + (size_t)m * DD + e);
    v.x += gv.x; v.y += gv.y; v.z += gv.z; v.w += gv.w;
  }
  size_t o = (size_t)rowi * DD + e;
  teF[o] = v.x; teF[o + 1] = v.y; teF[o + 2] = v.z; teF[o + 3] = v.w;
  teH[o] = (_Float16)v.x; teH[o + 1] = (_Float16)v.y;
  teH[o + 2] = (_Float16)v.z; teH[o + 3] = (_Float16)v.w;
}

// ---------------------------------------------------------------------------
// Attention per (b,h): 100x100 scores in LDS, softmax, att @ v. Writes f16 o
// (consumed only by the Wo WMMA GEMM).
// ---------------------------------------------------------------------------
__global__ void k_attn(const float* __restrict__ q, const float* __restrict__ k,
                       const float* __restrict__ v, _Float16* __restrict__ oH) {
  int b = blockIdx.x, h = blockIdx.y;
  __shared__ float sq[KK * 32], sk[KK * 32], sv[KK * 32];
  __shared__ float ss[KK * KK];
  int tid = threadIdx.x;
  for (int i = tid; i < KK * 32; i += blockDim.x) {
    int t = i >> 5, j = i & 31;
    size_t base = (size_t)(b * KK + t) * DD + h * 32 + j;
    sq[i] = q[base]; sk[i] = k[base]; sv[i] = v[base];
  }
  __syncthreads();
  const float scale = 0.17677669529663687f;  // 1/sqrt(32)
  for (int i = tid; i < KK * KK; i += blockDim.x) {
    int t = i / KK, s = i % KK;
    float acc = 0.f;
#pragma unroll 8
    for (int j = 0; j < 32; ++j) acc += sq[t * 32 + j] * sk[s * 32 + j];
    ss[i] = acc * scale;
  }
  __syncthreads();
  for (int t = tid; t < KK; t += blockDim.x) {
    float mx = -1e30f;
    for (int s = 0; s < KK; ++s) mx = fmaxf(mx, ss[t * KK + s]);
    float sum = 0.f;
    for (int s = 0; s < KK; ++s) { float e = __expf(ss[t * KK + s] - mx); ss[t * KK + s] = e; sum += e; }
    float inv = 1.f / sum;
    for (int s = 0; s < KK; ++s) ss[t * KK + s] *= inv;
  }
  __syncthreads();
  for (int i = tid; i < KK * 32; i += blockDim.x) {
    int t = i >> 5, j = i & 31;
    float acc = 0.f;
    for (int s = 0; s < KK; ++s) acc += ss[t * KK + s] * sv[s * 32 + j];
    oH[(size_t)(b * KK + t) * DD + h * 32 + j] = (_Float16)acc;
  }
}

// out = LayerNorm(A + f(Bv)), f = relu or identity; optional f16 copy.
__global__ void k_add_ln(const float* __restrict__ A, const float* __restrict__ Bv,
                         const float* __restrict__ g, const float* __restrict__ bta,
                         float* __restrict__ outF, _Float16* __restrict__ outH, int doRelu) {
  int row = blockIdx.x, d = threadIdx.x;  // 128 threads per row
  __shared__ float red[DD];
  float bv = Bv[(size_t)row * DD + d];
  if (doRelu) bv = fmaxf(bv, 0.f);
  float x = A[(size_t)row * DD + d] + bv;
  red[d] = x;
  __syncthreads();
  for (int s = 64; s > 0; s >>= 1) { if (d < s) red[d] += red[d + s]; __syncthreads(); }
  float mu = red[0] / (float)DD;
  __syncthreads();
  float c = x - mu;
  red[d] = c * c;
  __syncthreads();
  for (int s = 64; s > 0; s >>= 1) { if (d < s) red[d] += red[d + s]; __syncthreads(); }
  float var = red[0] / (float)DD;
  float y = c * rsqrtf(var + 1e-5f) * g[d] + bta[d];
  outF[(size_t)row * DD + d] = y;
  if (outH) outH[(size_t)row * DD + d] = (_Float16)y;
}

// scores[b,t] = lhs_proj[batch[b]] . x2[b,t]; scatter into -inf tensor.
__global__ void k_scores(const float* __restrict__ lhsProj, const int* __restrict__ batch,
                         const float* __restrict__ x2, const int* __restrict__ topk,
                         float* __restrict__ tlog) {
  int g = blockIdx.x * blockDim.x + threadIdx.x;
  if (g >= BB * KK) return;
  int b = g / KK;
  const float* lq = lhsProj + (size_t)batch[b] * DD;
  const float* xr = x2 + (size_t)g * DD;
  float s = 0.f;
#pragma unroll 8
  for (int d = 0; d < DD; ++d) s += lq[d] * xr[d];
  tlog[(size_t)b * NN_ + topk[g]] = s;
}

// ---------------------------------------------------------------------------
extern "C" void kernel_launch(void* const* d_in, const int* in_sizes, int n_in,
                              void* d_out, int out_size, void* d_ws, size_t ws_size,
                              hipStream_t stream) {
  const float* lhs   = (const float*)d_in[0];
  const float* gnn   = (const float*)d_in[1];
  const float* W     = (const float*)d_in[2];
  const float* projW = (const float*)d_in[3];
  const float* projb = (const float*)d_in[4];
  const float* headW = (const float*)d_in[5];
  const float* headb = (const float*)d_in[6];
  const float* offEW = (const float*)d_in[7];
  const float* offEb = (const float*)d_in[8];
  const float* offIW = (const float*)d_in[9];
  const float* offIb = (const float*)d_in[10];
  const float* Wq = (const float*)d_in[11]; const float* bq = (const float*)d_in[12];
  const float* Wk = (const float*)d_in[13]; const float* bk = (const float*)d_in[14];
  const float* Wv = (const float*)d_in[15]; const float* bv = (const float*)d_in[16];
  const float* Wo = (const float*)d_in[17]; const float* bo = (const float*)d_in[18];
  const float* ln1g = (const float*)d_in[19]; const float* ln1b = (const float*)d_in[20];
  const float* ffnW = (const float*)d_in[21]; const float* ffnb = (const float*)d_in[22];
  const float* ln2g = (const float*)d_in[23]; const float* ln2b = (const float*)d_in[24];
  const int* idgnnIdx = (const int*)d_in[25];
  const int* batch    = (const int*)d_in[26];
  (void)in_sizes; (void)n_in; (void)out_size; (void)ws_size;

  float* out    = (float*)d_out;
  float* embgnn = out;                                  // [B,N] f32
  float* tlog   = out + (size_t)BB * NN_;               // [B,N] f32
  int*   topk   = (int*)(out + 2 * (size_t)BB * NN_);   // [B,TOPK] i32

  char* base = (char*)d_ws; size_t off = 0;
  auto alloc = [&](size_t bytes) -> void* {
    void* p = base + off; off = (off + bytes + 255) & ~(size_t)255; return p;
  };
  float*    lhsProjF  = (float*)alloc((size_t)BB * DD * 4);
  _Float16* lhsProjH  = (_Float16*)alloc((size_t)BB * DD * 2);
  float*    offEmbRow = (float*)alloc(BB * 4);
  float*    offIdRow  = (float*)alloc(BB * 4);
  int*      node2m    = (int*)alloc((size_t)NN_ * 4);
  float*    teF  = (float*)alloc((size_t)BB * KK * DD * 4);
  _Float16* teH  = (_Float16*)alloc((size_t)BB * KK * DD * 2);
  float*    qF   = (float*)alloc((size_t)BB * KK * DD * 4);
  float*    kF   = (float*)alloc((size_t)BB * KK * DD * 4);
  float*    vF   = (float*)alloc((size_t)BB * KK * DD * 4);
  _Float16* oH   = (_Float16*)alloc((size_t)BB * KK * DD * 2);
  float*    opF  = (float*)alloc((size_t)BB * KK * DD * 4);
  float*    x1F  = (float*)alloc((size_t)BB * KK * DD * 4);
  _Float16* x1H  = (_Float16*)alloc((size_t)BB * KK * DD * 2);
  float*    ffF  = (float*)alloc((size_t)BB * KK * DD * 4);
  float*    x2F  = (float*)alloc((size_t)BB * KK * DD * 4);

  // 1) lhs_proj (+f16 copy) and per-row offsets
  k_lhs_proj<<<1, 128, 0, stream>>>(lhs, projW, projb, offEW, offEb, offIW, offIb,
                                    lhsProjF, lhsProjH, offEmbRow, offIdRow);

  // 2) dense [B,N] logits via WMMA (memory bound: 256 MB stream -> ~14 us floor)
  {
    int waves = (BB / 32) * (NN_ / 16);
    k_gemm_wmma<<<(waves + 7) / 8, 256, 0, stream>>>(lhsProjH, W, nullptr, embgnn,
                                                     BB, NN_, offEmbRow);
  }

  // 3) node->m reverse map + idgnn scatter into dense logits
  k_fill_i32<<<(NN_ + 255) / 256, 256, 0, stream>>>(node2m, -1, NN_);
  k_node2m_scatter<<<(MM + 255) / 256, 256, 0, stream>>>(idgnnIdx, node2m);
  k_idgnn<<<(MM + 255) / 256, 256, 0, stream>>>(lhs, gnn, headW, headb, offIdRow,
                                                idgnnIdx, batch, embgnn);

  // 4) exact per-row top-100 (radix select, rows live in L2)
  k_topk<<<BB, 1024, 0, stream>>>(embgnn, topk);

  // 5) gather final rhs embeddings for the top rows
  k_gather<<<(BB * KK * (DD / 4) + 255) / 256, 256, 0, stream>>>(W, gnn, node2m, topk,
                                                                 teF, teH);

  // 6-9) MAB transformer: q/k/v/o/ffn projections via the same WMMA GEMM
  {
    int waves = (BB * KK / 32) * (DD / 16);
    int blk = (waves + 7) / 8;
    k_gemm_wmma<<<blk, 256, 0, stream>>>(teH, Wq, bq, qF, BB * KK, DD, nullptr);
    k_gemm_wmma<<<blk, 256, 0, stream>>>(teH, Wk, bk, kF, BB * KK, DD, nullptr);
    k_gemm_wmma<<<blk, 256, 0, stream>>>(teH, Wv, bv, vF, BB * KK, DD, nullptr);
    k_attn<<<dim3(BB, HH), 128, 0, stream>>>(qF, kF, vF, oH);
    k_gemm_wmma<<<blk, 256, 0, stream>>>(oH, Wo, bo, opF, BB * KK, DD, nullptr);
    k_add_ln<<<BB * KK, 128, 0, stream>>>(teF, opF, ln1g, ln1b, x1F, x1H, 0);
    k_gemm_wmma<<<blk, 256, 0, stream>>>(x1H, ffnW, ffnb, ffF, BB * KK, DD, nullptr);
    k_add_ln<<<BB * KK, 128, 0, stream>>>(x1F, ffF, ln2g, ln2b, x2F, nullptr, 1);
  }

  // 10) transformer logits: -inf fill then scatter rerank scores
  k_fill_f32<<<2048, 256, 0, stream>>>(tlog, -__builtin_inff(), (long long)BB * NN_);
  k_scores<<<(BB * KK + 255) / 256, 256, 0, stream>>>(lhsProjF, batch, x2F, topk, tlog);
}
